// EncoderDecoder_15590731285096
// MI455X (gfx1250) — matrix operand order repaired and checked by reading, not compiled, not run on hardware
//
#include <hip/hip_runtime.h>
#include <math.h>

// ---------------------------------------------------------------------------
// EncoderDecoder stack for MI455X (gfx1250, wave32).
// All GEMMs run on v_wmma_f32_16x16x32_bf16 (fp32 accumulate); everything
// else (softmax over batch axis, Frobenius scale, LayerNorm, exact GELU)
// stays fp32. GEMM uses double-buffered LDS staging with a pipelined
// global->register->LDS copy so loads of tile k+1 overlap WMMAs on tile k.
// ---------------------------------------------------------------------------

typedef __attribute__((ext_vector_type(16))) __bf16 v16bf;
typedef __attribute__((ext_vector_type(8)))  float  v8f;

union BF16Frag { v16bf v; uint4 q[2]; };

#define TILE 128
#define KT 32
#define LDS_STRIDE 40   // bf16 per LDS row: 80B, multiple of 16B -> aligned b128 reads

__device__ __forceinline__ float gelu_exact(float x) {
    return 0.5f * x * (1.0f + erff(x * 0.70710678118654752f));
}

// ---------------------------------------------------------------------------
// Generic tiled WMMA GEMM:  C[M,N] = epilogue(A[M,K] @ B)  (optionally batched)
//   EPI: 0 = none, 1 = multiply by sumsq^{-1/4} (attention scale), 2 = bias+GELU
//   BT : 0 = B stored [K][N] row-major, 1 = B stored [N][K] (computes A*B^T)
// Block: 256 threads = 8 waves; block tile 128x128; wave tile 32x64 (2x4 WMMA).
// Double-buffered LDS, software-pipelined global loads.
// ---------------------------------------------------------------------------
template <int EPI, int BT>
__global__ __launch_bounds__(256) void gemm_wmma_bf16(
    const float* __restrict__ A, const float* __restrict__ B, float* __restrict__ C,
    int M, int N, int K, int lda, int ldb, int ldc,
    long long sA, long long sB, long long sC,
    const float* __restrict__ scalePtr, const float* __restrict__ bias)
{
    A += (long long)blockIdx.z * sA;
    B += (long long)blockIdx.z * sB;
    C += (long long)blockIdx.z * sC;

    const int tid  = threadIdx.x;
    const int lane = tid & 31;
    const int wave = tid >> 5;
    const int wm   = wave >> 1;     // 0..3 : 32-row strip
    const int wn   = wave & 1;      // 0..1 : 64-col strip
    const int lm   = lane & 15;
    const int hi   = lane >> 4;

    const int row0 = blockIdx.y * TILE;
    const int col0 = blockIdx.x * TILE;

    __shared__ __bf16 As[2][TILE * LDS_STRIDE];
    __shared__ __bf16 Bs[2][TILE * LDS_STRIDE];

    // Per-thread staging coordinates (fixed across K-steps).
    const int ar  = tid >> 3;           // row within 128-row tile (j adds 32)
    const int ac4 = (tid & 7) << 2;     // float4 column offset within 32-wide slab
    const int bk  = tid >> 5;           // k within slab for the transpose path (j adds 8)
    const int bn4 = (tid & 31) << 2;    // col within 128-col tile (transpose path)

    float4 aReg[4], bReg[4];

    auto loadTile = [&](int kk0) {
#pragma unroll
        for (int j = 0; j < 4; ++j)
            aReg[j] = *(const float4*)(A + (long long)(row0 + ar + 32 * j) * lda + kk0 + ac4);
        if (BT) {
#pragma unroll
            for (int j = 0; j < 4; ++j)
                bReg[j] = *(const float4*)(B + (long long)(col0 + ar + 32 * j) * ldb + kk0 + ac4);
        } else {
#pragma unroll
            for (int j = 0; j < 4; ++j)
                bReg[j] = *(const float4*)(B + (long long)(kk0 + bk + 8 * j) * ldb + col0 + bn4);
        }
    };

    auto storeTile = [&](int buf) {
#pragma unroll
        for (int j = 0; j < 4; ++j) {
            __bf16* d = &As[buf][(ar + 32 * j) * LDS_STRIDE + ac4];
            d[0] = (__bf16)aReg[j].x; d[1] = (__bf16)aReg[j].y;
            d[2] = (__bf16)aReg[j].z; d[3] = (__bf16)aReg[j].w;
        }
        if (BT) {
#pragma unroll
            for (int j = 0; j < 4; ++j) {
                __bf16* d = &Bs[buf][(ar + 32 * j) * LDS_STRIDE + ac4];
                d[0] = (__bf16)bReg[j].x; d[1] = (__bf16)bReg[j].y;
                d[2] = (__bf16)bReg[j].z; d[3] = (__bf16)bReg[j].w;
            }
        } else {
#pragma unroll
            for (int j = 0; j < 4; ++j) {
                const int kk = bk + 8 * j;
                Bs[buf][(bn4 + 0) * LDS_STRIDE + kk] = (__bf16)bReg[j].x;
                Bs[buf][(bn4 + 1) * LDS_STRIDE + kk] = (__bf16)bReg[j].y;
                Bs[buf][(bn4 + 2) * LDS_STRIDE + kk] = (__bf16)bReg[j].z;
                Bs[buf][(bn4 + 3) * LDS_STRIDE + kk] = (__bf16)bReg[j].w;
            }
        }
    };

    v8f zero = {};
    v8f acc[2][4];
#pragma unroll
    for (int mt = 0; mt < 2; ++mt)
#pragma unroll
        for (int nt = 0; nt < 4; ++nt) acc[mt][nt] = zero;

    loadTile(0);
    int buf = 0;

    for (int kk0 = 0; kk0 < K; kk0 += KT) {
        storeTile(buf);
        __syncthreads();
        if (kk0 + KT < K) {
            loadTile(kk0 + KT);                 // overlap with WMMAs below
            if (kk0 + 2 * KT < K) {             // warm L2 two slabs ahead
                __builtin_prefetch(A + (long long)(row0 + ar) * lda + kk0 + 2 * KT, 0, 0);
                __builtin_prefetch(BT ? (B + (long long)(col0 + ar) * ldb + kk0 + 2 * KT)
                                      : (B + (long long)(kk0 + 2 * KT + bk) * ldb + col0 + bn4),
                                   0, 0);
            }
        }

        // ---- fragments per ISA 7.12.2 16-bit layouts ----
        BF16Frag afr[2], bfr[4];
        const int akb = hi ? 8 : 0;             // lanes 16-31: K{8-15,24-31}
#pragma unroll
        for (int mt = 0; mt < 2; ++mt) {
            const __bf16* p = &As[buf][(wm * 32 + mt * 16 + lm) * LDS_STRIDE + akb];
            afr[mt].q[0] = *(const uint4*)(p);
            afr[mt].q[1] = *(const uint4*)(p + 16);
        }
        const int bkb = hi ? 8 : 0;             // lanes 16-31: K{8-15,24-31}
#pragma unroll
        for (int nt = 0; nt < 4; ++nt) {
            const __bf16* p = &Bs[buf][(wn * 64 + nt * 16 + lm) * LDS_STRIDE + bkb];
            bfr[nt].q[0] = *(const uint4*)(p);
            bfr[nt].q[1] = *(const uint4*)(p + 16);
        }
#pragma unroll
        for (int mt = 0; mt < 2; ++mt)
#pragma unroll
            for (int nt = 0; nt < 4; ++nt)
                acc[mt][nt] = __builtin_amdgcn_wmma_f32_16x16x32_bf16(
                    false, afr[mt].v, false, bfr[nt].v,
                    (short)0, acc[mt][nt], false, false);

        buf ^= 1;
    }

    // ---- epilogue ----
    float mul = 1.0f;
    if (EPI == 1) mul = rsqrtf(sqrtf(*scalePtr));   // 1 / (||x||_F)^{1/2} = sumsq^{-1/4}
#pragma unroll
    for (int mt = 0; mt < 2; ++mt) {
#pragma unroll
        for (int nt = 0; nt < 4; ++nt) {
            const int col = col0 + wn * 64 + nt * 16 + lm;
            const int rb  = row0 + wm * 32 + mt * 16 + hi * 8;
            float bv = (EPI == 2) ? bias[col] : 0.0f;
#pragma unroll
            for (int r = 0; r < 8; ++r) {
                float v = acc[mt][nt][r];
                if (EPI == 1) v *= mul;
                if (EPI == 2) v = gelu_exact(v + bv);
                C[(long long)(rb + r) * ldc + col] = v;
            }
        }
    }
}

// ---------------------------------------------------------------------------
// Frobenius sum-of-squares reduction (scale = sumsq^{1/4} applied in GEMM).
// ---------------------------------------------------------------------------
__global__ __launch_bounds__(256) void reduce_sumsq(
    const float* __restrict__ x, long long n, float* __restrict__ out)
{
    __shared__ float red[256];
    float s = 0.f;
    for (long long i = (long long)blockIdx.x * 256 + threadIdx.x; i < n;
         i += (long long)gridDim.x * 256) {
        float v = x[i];
        s += v * v;
    }
    red[threadIdx.x] = s;
    __syncthreads();
    for (int o = 128; o > 0; o >>= 1) {
        if (threadIdx.x < (unsigned)o) red[threadIdx.x] += red[threadIdx.x + o];
        __syncthreads();
    }
    if (threadIdx.x == 0) atomicAdd(out, red[0]);
}

// ---------------------------------------------------------------------------
// softmax over the BATCH axis (axis=0 of a[8,S,S]), per (s,t) position.
// ---------------------------------------------------------------------------
__global__ void softmax_over_batch(float* __restrict__ a, int SS)
{
    int i = blockIdx.x * blockDim.x + threadIdx.x;
    if (i >= SS) return;
    float v[8];
    float m = -3.402823466e+38f;
#pragma unroll
    for (int b = 0; b < 8; ++b) {
        v[b] = a[(long long)b * SS + i];
        m = fmaxf(m, v[b]);
    }
    float s = 0.f;
#pragma unroll
    for (int b = 0; b < 8; ++b) { v[b] = expf(v[b] - m); s += v[b]; }
    float inv = 1.f / s;
#pragma unroll
    for (int b = 0; b < 8; ++b) a[(long long)b * SS + i] = v[b] * inv;
}

// ---------------------------------------------------------------------------
// Row LayerNorm: one 256-thread block per row, row cached in LDS (C <= 4096).
// ---------------------------------------------------------------------------
__global__ __launch_bounds__(256) void layernorm_rows(
    const float* __restrict__ x, float* __restrict__ y,
    const float* __restrict__ g, const float* __restrict__ b, int C)
{
    __shared__ float buf[4096];
    __shared__ float red[256];
    const long long row = blockIdx.x;
    const float* xr = x + row * C;
    float s = 0.f, s2 = 0.f;
    for (int c = threadIdx.x; c < C; c += 256) {
        float v = xr[c];
        buf[c] = v;
        s += v; s2 += v * v;
    }
    red[threadIdx.x] = s;
    __syncthreads();
    for (int o = 128; o > 0; o >>= 1) {
        if (threadIdx.x < (unsigned)o) red[threadIdx.x] += red[threadIdx.x + o];
        __syncthreads();
    }
    const float mean = red[0] / (float)C;
    __syncthreads();
    red[threadIdx.x] = s2;
    __syncthreads();
    for (int o = 128; o > 0; o >>= 1) {
        if (threadIdx.x < (unsigned)o) red[threadIdx.x] += red[threadIdx.x + o];
        __syncthreads();
    }
    const float var = red[0] / (float)C - mean * mean;
    const float inv = rsqrtf(var + 1e-5f);
    float* yr = y + row * C;
    for (int c = threadIdx.x; c < C; c += 256)
        yr[c] = (buf[c] - mean) * inv * g[c] + b[c];
}

// ---------------------------------------------------------------------------
// Host orchestration. d_in order: x, then per encoder {Wq,Wk,g0,b0,W,b,g1,b1},
// then decoder {W,b,g1,b1}.
// ---------------------------------------------------------------------------
extern "C" void kernel_launch(void* const* d_in, const int* in_sizes, int n_in,
                              void* d_out, int out_size, void* d_ws, size_t ws_size,
                              hipStream_t stream)
{
    (void)in_sizes; (void)n_in; (void)out_size; (void)ws_size;

    const int Bn = 8, S = 1024, Mq = 128;
    const long long BS = (long long)Bn * S;                 // 8192 rows

    float* ws     = (float*)d_ws;
    float* sumsq  = ws;                                     // 1 float (padded)
    float* qbuf   = ws + 64;                                // 8192*128
    float* kbuf   = qbuf   + BS * Mq;                       // 8192*128
    float* logits = kbuf   + BS * Mq;                       // 8*1024*1024
    float* vatt   = logits + (long long)Bn * S * S;         // up to 8192*2048
    float* hA     = vatt   + BS * 2048;                     // up to 8192*4096
    float* hB     = hA     + BS * 4096;                     // up to 8192*4096

    const int ci_arr[4] = {256, 512, 1024, 2048};
    const int co_arr[4] = {512, 1024, 2048, 4096};
    float* outs[4] = {hA, hB, hA, hB};

    const float* X = (const float*)d_in[0];

    for (int l = 0; l < 4; ++l) {
        const int ci = ci_arr[l], co = co_arr[l];
        const float* Wq = (const float*)d_in[1 + 8 * l + 0];
        const float* Wk = (const float*)d_in[1 + 8 * l + 1];
        const float* g0 = (const float*)d_in[1 + 8 * l + 2];
        const float* b0 = (const float*)d_in[1 + 8 * l + 3];
        const float* W  = (const float*)d_in[1 + 8 * l + 4];
        const float* bb = (const float*)d_in[1 + 8 * l + 5];
        const float* g1 = (const float*)d_in[1 + 8 * l + 6];
        const float* b1 = (const float*)d_in[1 + 8 * l + 7];
        float* Y = outs[l];

        // scale = sqrt(||X||_F)  (sum of squares on device, sqrt.sqrt in epilogue)
        hipMemsetAsync(sumsq, 0, sizeof(float), stream);
        reduce_sumsq<<<512, 256, 0, stream>>>(X, BS * ci, sumsq);

        // q = X @ Wq, k = X @ Wk      [8192,ci]x[ci,128]
        gemm_wmma_bf16<0, 0><<<dim3(Mq / 128, (int)(BS / 128), 1), 256, 0, stream>>>(
            X, Wq, qbuf, (int)BS, Mq, ci, ci, Mq, Mq, 0, 0, 0, nullptr, nullptr);
        gemm_wmma_bf16<0, 0><<<dim3(Mq / 128, (int)(BS / 128), 1), 256, 0, stream>>>(
            X, Wk, kbuf, (int)BS, Mq, ci, ci, Mq, Mq, 0, 0, 0, nullptr, nullptr);

        // logits_b = (q_b @ k_b^T) * sumsq^{-1/4}   batched over 8
        gemm_wmma_bf16<1, 1><<<dim3(S / 128, S / 128, Bn), 256, 0, stream>>>(
            qbuf, kbuf, logits, S, S, Mq, Mq, Mq, S,
            (long long)S * Mq, (long long)S * Mq, (long long)S * S, sumsq, nullptr);

        // softmax over the batch axis
        softmax_over_batch<<<(S * S + 255) / 256, 256, 0, stream>>>(logits, S * S);

        // v_att_b = a_b @ X_b        [1024,1024]x[1024,ci], batched over 8
        gemm_wmma_bf16<0, 0><<<dim3(ci / 128, S / 128, Bn), 256, 0, stream>>>(
            logits, X, vatt, S, ci, S, S, ci, ci,
            (long long)S * S, (long long)S * ci, (long long)S * ci, nullptr, nullptr);

        // LayerNorm(v_att; g0,b0) in place
        layernorm_rows<<<(int)BS, 256, 0, stream>>>(vatt, vatt, g0, b0, ci);

        // Y = LayerNorm(GELU(v_att @ W + b); g1,b1)
        gemm_wmma_bf16<2, 0><<<dim3(co / 128, (int)(BS / 128), 1), 256, 0, stream>>>(
            vatt, W, Y, (int)BS, co, ci, ci, co, co, 0, 0, 0, nullptr, bb);
        layernorm_rows<<<(int)BS, 256, 0, stream>>>(Y, Y, g1, b1, co);

        X = Y;
    }

    // decoder: out = LayerNorm(GELU(X @ W + b))   [8192,4096]x[4096,2048]
    const float* Wd  = (const float*)d_in[33];
    const float* bd  = (const float*)d_in[34];
    const float* gd  = (const float*)d_in[35];
    const float* bd1 = (const float*)d_in[36];
    gemm_wmma_bf16<2, 0><<<dim3(2048 / 128, (int)(BS / 128), 1), 256, 0, stream>>>(
        X, Wd, vatt, (int)BS, 2048, 4096, 4096, 2048, 2048, 0, 0, 0, nullptr, bd);
    layernorm_rows<<<(int)BS, 256, 0, stream>>>(vatt, (float*)d_out, gd, bd1, 2048);
}
